// SimpleMambaDriverDistraction_3435973836968
// MI455X (gfx1250) — matrix operand order repaired and checked by reading, not compiled, not run on hardware
//
#include <hip/hip_runtime.h>
#include <hip/hip_bf16.h>

// ---------------- model constants ----------------
#define BATCH   32
#define IMG     224
#define EDIM    256
#define DIN     512
#define DSTATE  64
#define DCONV   4
#define DRANK   16
#define NBLK    6
#define SEQL    196          // (224/16)^2
#define NCLS    10
#define EPSV    1e-5f
#define SCH     14           // scan chunk: 196 = 14 * 14

// ---------------- WMMA types ----------------
typedef __attribute__((ext_vector_type(16))) __bf16          v16bf;
typedef __attribute__((ext_vector_type(16))) unsigned short  v16u;
typedef __attribute__((ext_vector_type(8)))  float           v8f;

#define PA 40   // LDS row pitch (halves): 32 data + 8 pad; 80B rows keep 16B alignment
#define PB 40

__device__ __forceinline__ unsigned short f2bf(float f) {
    unsigned u = __builtin_bit_cast(unsigned, f);
    u += 0x7FFFu + ((u >> 16) & 1u);      // round-to-nearest-even
    return (unsigned short)(u >> 16);
}
__device__ __forceinline__ unsigned pack2(float a, float b) {
    return (unsigned)f2bf(a) | ((unsigned)f2bf(b) << 16);
}
__device__ __forceinline__ float sigmoidf_(float x) { return 1.f / (1.f + __expf(-x)); }
__device__ __forceinline__ float siluf_(float x)    { return x * sigmoidf_(x); }
__device__ __forceinline__ float softplusf_(float x){ return (x > 20.f) ? x : log1pf(__expf(x)); }

// A fragment: 16x32 bf16, lane L holds row M=L&15; K pattern per ISA table
__device__ __forceinline__ v16bf fragA(const unsigned short* s, int rowBase, int lane) {
    int m  = rowBase + (lane & 15);
    int kh = (lane >= 16) ? 8 : 0;
    v16u t;
#pragma unroll
    for (int j = 0; j < 8; ++j) {
        int k = 2 * j + ((j >= 4) ? 8 : 0) + kh;
        t[2 * j]     = s[m * PA + k];
        t[2 * j + 1] = s[m * PA + k + 1];
    }
    return __builtin_bit_cast(v16bf, t);
}
// B fragment: 32x16 bf16, lane L holds column N=L&15; K pattern per ISA table (same as A)
__device__ __forceinline__ v16bf fragB(const unsigned short* s, int rowBase, int lane) {
    int n  = rowBase + (lane & 15);
    int kh = (lane >= 16) ? 8 : 0;
    v16u t;
#pragma unroll
    for (int j = 0; j < 8; ++j) {
        int k = 2 * j + ((j >= 4) ? 8 : 0) + kh;
        t[2 * j]     = s[n * PB + k];
        t[2 * j + 1] = s[n * PB + k + 1];
    }
    return __builtin_bit_cast(v16bf, t);
}

// ============================================================
// Generic bf16 WMMA GEMM:  C[m,n] = act( sum_k A[m,k]*B[n,k] + bias[n] + res[m,n] )
//   A [M x K] lda, B [N x K] ldb (C = A*B^T). Double-buffered LDS: 1 barrier / K-step,
//   next-step global loads overlap WMMA. Branchless guards (clamp + select).
//   KG requires per-thread 8/16-wide K-segments uniformly valid (true for K=16).
// ============================================================
template<bool MG, bool NG, bool KG>
__global__ __launch_bounds__(256) void gemm_bf16_wmma(
    const float* __restrict__ A, int lda,
    const float* __restrict__ Bw, int ldb,
    float* __restrict__ C, int ldc,
    const float* __restrict__ bias, const float* __restrict__ res,
    int M, int N, int K, int act)
{
    __shared__ alignas(16) unsigned short As[2][64 * PA];
    __shared__ alignas(16) unsigned short Bs[2][128 * PB];

    int t = threadIdx.x, lane = t & 31, wid = t >> 5;
    int wm = wid & 1, wn = wid >> 1;
    int m0 = blockIdx.x * 64, n0 = blockIdx.y * 128;

    int ar = t >> 2, aks = (t & 3) * 8;        // A: 4 threads/row, 8 elems each
    int br = t >> 1, bks = (t & 1) * 16;       // B: 2 threads/row, 16 elems each
    int am = m0 + ar, bn_ = n0 + br;
    int arow = MG ? (am < M ? am : M - 1) : am;
    int brow = NG ? (bn_ < N ? bn_ : N - 1) : bn_;

    float4 a0, a1, b0, b1, b2, b3;

    auto loadAB = [&](int kk) {
        {
            int ka = kk + aks;
            int ks = KG ? (ka < K ? ka : 0) : ka;          // uniform per thread
            const float* ap = A + (long)arow * lda + ks;
            a0 = *(const float4*)ap;
            a1 = *(const float4*)(ap + 4);
            __builtin_prefetch(ap + 64, 0, 3);
            if (MG || KG) {
                bool ok = (!MG || am < M) && (!KG || ka < K);
                if (!ok) { a0 = make_float4(0, 0, 0, 0); a1 = make_float4(0, 0, 0, 0); }
            }
        }
        {
            int kb = kk + bks;
            int ks = KG ? (kb < K ? kb : 0) : kb;
            const float* bp = Bw + (long)brow * ldb + ks;
            b0 = *(const float4*)bp;
            b1 = *(const float4*)(bp + 4);
            b2 = *(const float4*)(bp + 8);
            b3 = *(const float4*)(bp + 12);
            __builtin_prefetch(bp + 64, 0, 3);
            if (NG || KG) {
                bool ok = (!NG || bn_ < N) && (!KG || kb < K);
                if (!ok) {
                    b0 = make_float4(0, 0, 0, 0); b1 = make_float4(0, 0, 0, 0);
                    b2 = make_float4(0, 0, 0, 0); b3 = make_float4(0, 0, 0, 0);
                }
            }
        }
    };
    auto storeAB = [&](int buf) {
        uint4 pk;
        pk.x = pack2(a0.x, a0.y); pk.y = pack2(a0.z, a0.w);
        pk.z = pack2(a1.x, a1.y); pk.w = pack2(a1.z, a1.w);
        *(uint4*)&As[buf][ar * PA + aks] = pk;
        uint4 p0, p1;
        p0.x = pack2(b0.x, b0.y); p0.y = pack2(b0.z, b0.w);
        p0.z = pack2(b1.x, b1.y); p0.w = pack2(b1.z, b1.w);
        p1.x = pack2(b2.x, b2.y); p1.y = pack2(b2.z, b2.w);
        p1.z = pack2(b3.x, b3.y); p1.w = pack2(b3.z, b3.w);
        *(uint4*)&Bs[buf][br * PB + bks]     = p0;
        *(uint4*)&Bs[buf][br * PB + bks + 8] = p1;
    };

    v8f acc[2][2] = {};

    loadAB(0);
    storeAB(0);
    __syncthreads();

    int buf = 0;
    for (int kk = 0; kk < K; kk += 32) {
        bool more = (kk + 32) < K;
        if (more) loadAB(kk + 32);            // overlap with WMMA below

        v16bf af[2], bfv[2];
#pragma unroll
        for (int i = 0; i < 2; ++i) af[i]  = fragA(&As[buf][0], wm * 32 + i * 16, lane);
#pragma unroll
        for (int j = 0; j < 2; ++j) bfv[j] = fragB(&Bs[buf][0], wn * 32 + j * 16, lane);
#pragma unroll
        for (int i = 0; i < 2; ++i)
#pragma unroll
            for (int j = 0; j < 2; ++j)
                acc[i][j] = __builtin_amdgcn_wmma_f32_16x16x32_bf16(
                    false, af[i], false, bfv[j], (short)0, acc[i][j], false, false);

        if (more) storeAB(buf ^ 1);
        buf ^= 1;
        __syncthreads();
    }

    int mo = (lane >= 16) ? 8 : 0, nc = lane & 15;
#pragma unroll
    for (int i = 0; i < 2; ++i)
#pragma unroll
        for (int j = 0; j < 2; ++j)
#pragma unroll
            for (int r = 0; r < 8; ++r) {
                int gm = m0 + wm * 32 + i * 16 + mo + r;
                int gn = n0 + wn * 32 + j * 16 + nc;
                if ((!MG || gm < M) && (!NG || gn < N)) {
                    float v = acc[i][j][r];
                    if (bias) v += bias[gn];
                    if (res)  v += res[(long)gm * ldc + gn];
                    if (act == 1) v = softplusf_(v);
                    C[(long)gm * ldc + gn] = v;
                }
            }
}

// ============================================================
// Implicit-GEMM 3x3 conv on ZERO-PADDED NCHW input + bias + BN + ReLU, bf16 WMMA.
//   in: [BATCH, Ci, IHP, IWP] already padded -> no bounds checks in the gather.
//   k -> input offset precomputed in LDS table: koff[k] = (ci*IHP+ky)*IWP+kx (-1 = K tail).
//   M = BATCH*OH*OW (multiple of 64 here), N = Co, K = Ci*9.
// ============================================================
template<bool NG, bool KG>
__global__ __launch_bounds__(256) void conv3x3_bn_relu_wmma(
    const float* __restrict__ in, const float* __restrict__ w,
    float* __restrict__ out,
    int Ci, int IHP, int IWP, int Co, int OH, int OW, int stride,
    const float* __restrict__ bng, const float* __restrict__ bnb,
    const float* __restrict__ bnm, const float* __restrict__ bnv,
    const float* __restrict__ cbias)
{
    __shared__ alignas(16) unsigned short As[64 * PA];
    __shared__ alignas(16) unsigned short Bs[128 * PB];
    __shared__ alignas(16) int koff[1152];   // max K = 128*9

    const int K = Ci * 9;
    const int OHW = OH * OW;

    // build k -> padded-input-offset table (once per block)
    {
        int K32 = (K + 31) & ~31;
        for (int k = threadIdx.x; k < K32; k += 256) {
            int kc = (k < K) ? k : 0;
            int ci = kc / 9; int r9 = kc - ci * 9; int ky = r9 / 3; int kx = r9 - ky * 3;
            int off = (ci * IHP + ky) * IWP + kx;
            koff[k] = (k < K) ? off : -1;
        }
    }

    int t = threadIdx.x, lane = t & 31, wid = t >> 5;
    int wm = wid & 1, wn = wid >> 1;
    int m0 = blockIdx.x * 64, n0 = blockIdx.y * 128;

    int ar = t >> 2, aks = (t & 3) * 8;
    int am = m0 + ar;                                  // am < M (M % 64 == 0)
    int ab = am / OHW; int rem = am - ab * OHW;
    int aoy = rem / OW, aox = rem - aoy * OW;
    int pixbase = (ab * Ci * IHP + aoy * stride) * IWP + aox * stride;  // fits int32
    int br = t >> 1, bn_ = n0 + br, bks = (t & 1) * 16;
    int brow = NG ? (bn_ < Co ? bn_ : Co - 1) : bn_;

    __syncthreads();   // table ready

    v8f acc[2][2] = {};

    for (int kk = 0; kk < K; kk += 32) {
        // ---- A: table-driven gather from padded input (no bounds checks) ----
        {
            int4 o0 = *(const int4*)&koff[kk + aks];
            int4 o1 = *(const int4*)&koff[kk + aks + 4];
            int oo[8] = { o0.x, o0.y, o0.z, o0.w, o1.x, o1.y, o1.z, o1.w };
            float va[8];
#pragma unroll
            for (int i = 0; i < 8; ++i) {
                int o = oo[i];
                if (KG) {
                    bool ok = o >= 0;
                    float v = in[pixbase + (ok ? o : 0)];
                    va[i] = ok ? v : 0.f;
                } else {
                    va[i] = in[pixbase + o];
                }
            }
            uint4 pk;
            pk.x = pack2(va[0], va[1]); pk.y = pack2(va[2], va[3]);
            pk.z = pack2(va[4], va[5]); pk.w = pack2(va[6], va[7]);
            *(uint4*)&As[ar * PA + aks] = pk;
        }
        // ---- B: weights ----
        if (KG) {
#pragma unroll
            for (int i = 0; i < 16; ++i) {
                int k = kk + bks + i;
                int kc = k < K ? k : K - 1;
                float v = w[brow * K + kc];
                bool ok = (!NG || bn_ < Co) && (k < K);
                Bs[br * PB + bks + i] = f2bf(ok ? v : 0.f);
            }
        } else {
            const float* bp = w + (long)brow * K + kk + bks;
            float4 v0 = *(const float4*)bp;
            float4 v1 = *(const float4*)(bp + 4);
            float4 v2 = *(const float4*)(bp + 8);
            float4 v3 = *(const float4*)(bp + 12);
            __builtin_prefetch(bp + 32, 0, 3);
            bool ok = (!NG || bn_ < Co);
            uint4 p0, p1;
            p0.x = pack2(v0.x, v0.y); p0.y = pack2(v0.z, v0.w);
            p0.z = pack2(v1.x, v1.y); p0.w = pack2(v1.z, v1.w);
            p1.x = pack2(v2.x, v2.y); p1.y = pack2(v2.z, v2.w);
            p1.z = pack2(v3.x, v3.y); p1.w = pack2(v3.z, v3.w);
            if (NG && !ok) {
                p0.x = 0; p0.y = 0; p0.z = 0; p0.w = 0;
                p1.x = 0; p1.y = 0; p1.z = 0; p1.w = 0;
            }
            *(uint4*)&Bs[br * PB + bks]     = p0;
            *(uint4*)&Bs[br * PB + bks + 8] = p1;
        }
        __syncthreads();

        v16bf af[2], bfv[2];
#pragma unroll
        for (int i = 0; i < 2; ++i) af[i]  = fragA(As, wm * 32 + i * 16, lane);
#pragma unroll
        for (int j = 0; j < 2; ++j) bfv[j] = fragB(Bs, wn * 32 + j * 16, lane);
#pragma unroll
        for (int i = 0; i < 2; ++i)
#pragma unroll
            for (int j = 0; j < 2; ++j)
                acc[i][j] = __builtin_amdgcn_wmma_f32_16x16x32_bf16(
                    false, af[i], false, bfv[j], (short)0, acc[i][j], false, false);
        __syncthreads();
    }

    int mo = (lane >= 16) ? 8 : 0, nc = lane & 15;
#pragma unroll
    for (int i = 0; i < 2; ++i)
#pragma unroll
        for (int j = 0; j < 2; ++j)
#pragma unroll
            for (int r = 0; r < 8; ++r) {
                int gm = m0 + wm * 32 + i * 16 + mo + r;
                int gn = n0 + wn * 32 + j * 16 + nc;
                if (!NG || gn < Co) {
                    int b_ = gm / OHW; int rm = gm - b_ * OHW;
                    int oy = rm / OW; int ox = rm - oy * OW;
                    float scale = bng[gn] * rsqrtf(bnv[gn] + EPSV);
                    float shift = (cbias[gn] - bnm[gn]) * scale + bnb[gn];
                    float v = fmaxf(acc[i][j][r] * scale + shift, 0.f);
                    out[((b_ * Co + gn) * OH + oy) * OW + ox] = v;
                }
            }
}

// ============================================================ small kernels
// zero-pad NCHW: out [B,C,IH+2p,IW+2p] <- in [B,C,IH,IW]
__global__ void pad_nchw_k(const float* __restrict__ in, float* __restrict__ out,
                           int C, int IH, int IW, int p)
{
    int IHP = IH + 2 * p, IWP = IW + 2 * p;
    long idx = (long)blockIdx.x * blockDim.x + threadIdx.x;
    long total = (long)BATCH * C * IHP * IWP;
    if (idx >= total) return;
    int xp = idx % IWP; long t = idx / IWP;
    int yp = t % IHP; t /= IHP;
    int c = t % C; int b = t / C;
    int iy = yp - p, ix = xp - p;
    bool ok = ((unsigned)iy < (unsigned)IH) && ((unsigned)ix < (unsigned)IW);
    int src = ok ? (((b * C + c) * IH + iy) * IW + ix) : 0;
    float v = in[src];
    out[idx] = ok ? v : 0.f;
}

__global__ void maxpool_k(const float* __restrict__ in, float* __restrict__ out,
                          int C, int IH, int IW, int OH, int OW, int kk, int s, int p)
{
    long idx = (long)blockIdx.x * blockDim.x + threadIdx.x;
    long total = (long)BATCH * C * OH * OW;
    if (idx >= total) return;
    int ox = idx % OW; long t = idx / OW;
    int oy = t % OH; t /= OH;
    int c = t % C; int b = t / C;
    float m = -INFINITY;
    for (int ky = 0; ky < kk; ++ky) {
        int iy = oy * s - p + ky;
        for (int kx = 0; kx < kk; ++kx) {
            int ix = ox * s - p + kx;
            bool ok = ((unsigned)iy < (unsigned)IH) && ((unsigned)ix < (unsigned)IW);
            int addr = ok ? (((b * C + c) * IH + iy) * IW + ix) : 0;
            float v = in[addr];
            m = fmaxf(m, ok ? v : -INFINITY);
        }
    }
    out[idx] = m;
}

__global__ void nchw_to_tokens_k(const float* __restrict__ in, float* __restrict__ out)
{
    long idx = (long)blockIdx.x * blockDim.x + threadIdx.x;
    long total = (long)BATCH * EDIM * SEQL;
    if (idx >= total) return;
    int p = idx % SEQL; long t = idx / SEQL;
    int e = t % EDIM; int b = t / EDIM;
    out[((long)b * SEQL + p) * EDIM + e] = in[idx];
}

__global__ __launch_bounds__(256) void layernorm_k(
    const float* __restrict__ x, const float* __restrict__ g,
    const float* __restrict__ bb, float* __restrict__ out)
{
    __shared__ float red[256];
    int row = blockIdx.x, t = threadIdx.x;
    float v = x[(long)row * EDIM + t];
    red[t] = v; __syncthreads();
    for (int s = 128; s > 0; s >>= 1) { if (t < s) red[t] += red[t + s]; __syncthreads(); }
    float mu = red[0] * (1.f / EDIM);
    __syncthreads();
    float d = v - mu;
    red[t] = d * d; __syncthreads();
    for (int s = 128; s > 0; s >>= 1) { if (t < s) red[t] += red[t + s]; __syncthreads(); }
    float var = red[0] * (1.f / EDIM);
    out[(long)row * EDIM + t] = d * rsqrtf(var + EPSV) * g[t] + bb[t];
}

// depthwise causal conv1d (DC=4) over tokens + silu;  xi = xz[:, :DIN] (row stride 2*DIN)
__global__ void dwconv_silu_k(const float* __restrict__ xz, const float* __restrict__ cw,
                              const float* __restrict__ cb, float* __restrict__ xc)
{
    long idx = (long)blockIdx.x * blockDim.x + threadIdx.x;
    long total = (long)BATCH * SEQL * DIN;
    if (idx >= total) return;
    int d = idx % DIN; long t = idx / DIN;
    int l = t % SEQL; int b = t / SEQL;
    float acc = cb[d];
#pragma unroll
    for (int j = 0; j < DCONV; ++j) {
        int ls = l - (DCONV - 1) + j;
        bool ok = ls >= 0;
        long addr = ok ? (((long)b * SEQL + ls) * (2 * DIN) + d) : 0;
        float v = xz[addr];
        acc += cw[d * DCONV + j] * (ok ? v : 0.f);
    }
    xc[idx] = siluf_(acc);
}

// ============================================================
// Fused selective scan, chunked: state h[64] + A row live in VGPRs per (b,d) thread.
// B/C vectors staged SCH=14 steps per barrier (28 barriers total instead of 392);
// dt/xc/z software-pipelined one step ahead to hide L2 latency behind the exp chain;
// 4-way split accumulator breaks the h.C reduction dependency chain.
// Epilogue fused: y = (h.C + d_skip*xc) * silu(z)
// ============================================================
__global__ __launch_bounds__(128) void ssm_scan_fused(
    const float* __restrict__ dtb,   // [B*L, DIN]
    const float* __restrict__ dbl,   // [B*L, 144]; Bvec at +DRANK, Cvec at +DRANK+DSTATE
    const float* __restrict__ xc,    // [B*L, DIN]
    const float* __restrict__ xz,    // [B*L, 2*DIN]; z at +DIN
    const float* __restrict__ alog,  // [DIN, DSTATE]
    const float* __restrict__ dskip, // [DIN]
    float* __restrict__ y)           // [B*L, DIN]
{
    __shared__ float sBC[SCH * 128];     // per step: [0..63]=B, [64..127]=C
    int tid = threadIdx.x;
    int d = blockIdx.x * 128 + tid;
    int b = blockIdx.y;
    float Aa[DSTATE], h[DSTATE];
#pragma unroll
    for (int s = 0; s < DSTATE; ++s) { Aa[s] = -__expf(alog[d * DSTATE + s]); h[s] = 0.f; }
    float dsk = dskip[d];
    const long rowb = (long)b * SEQL;

    // pipeline prologue: per-step scalars for l = 0
    float cdt = dtb[rowb * DIN + d];
    float cxc = xc[rowb * DIN + d];
    float cz  = xz[rowb * (2 * DIN) + DIN + d];

    for (int chunk = 0; chunk < SEQL; chunk += SCH) {
        // stage B/C for SCH steps (coalesced: 128 consecutive floats per row)
#pragma unroll
        for (int i = 0; i < SCH; ++i)
            sBC[i * 128 + tid] = dbl[(rowb + chunk + i) * 144 + DRANK + tid];
        __syncthreads();

        for (int lc = 0; lc < SCH; ++lc) {
            int l = chunk + lc;
            int ln = (l + 1 < SEQL) ? l + 1 : l;          // prefetch next step's scalars
            float ndt = dtb[(rowb + ln) * DIN + d];
            float nxc = xc[(rowb + ln) * DIN + d];
            float nz  = xz[(rowb + ln) * (2 * DIN) + DIN + d];

            const float* bc = &sBC[lc * 128];
            float dx = cdt * cxc;
            float d0 = 0.f, d1 = 0.f, d2 = 0.f, d3 = 0.f;
#pragma unroll
            for (int s = 0; s < DSTATE; s += 4) {
                h[s]     = __expf(cdt * Aa[s])     * h[s]     + dx * bc[s];
                h[s + 1] = __expf(cdt * Aa[s + 1]) * h[s + 1] + dx * bc[s + 1];
                h[s + 2] = __expf(cdt * Aa[s + 2]) * h[s + 2] + dx * bc[s + 2];
                h[s + 3] = __expf(cdt * Aa[s + 3]) * h[s + 3] + dx * bc[s + 3];
                d0 += h[s]     * bc[64 + s];
                d1 += h[s + 1] * bc[64 + s + 1];
                d2 += h[s + 2] * bc[64 + s + 2];
                d3 += h[s + 3] * bc[64 + s + 3];
            }
            float dot = (d0 + d1) + (d2 + d3);
            y[(rowb + l) * DIN + d] = (dot + dsk * cxc) * siluf_(cz);
            cdt = ndt; cxc = nxc; cz = nz;
        }
        __syncthreads();
    }
}

__global__ void meanpool_k(const float* __restrict__ x, float* __restrict__ out)
{
    int idx = blockIdx.x * blockDim.x + threadIdx.x;
    if (idx >= BATCH * EDIM) return;
    int e = idx % EDIM, b = idx / EDIM;
    float s = 0.f;
    for (int l = 0; l < SEQL; ++l) s += x[((long)b * SEQL + l) * EDIM + e];
    out[idx] = s * (1.f / SEQL);
}

// ============================================================ launcher
extern "C" void kernel_launch(void* const* d_in, const int* in_sizes, int n_in,
                              void* d_out, int out_size, void* d_ws, size_t ws_size,
                              hipStream_t stream)
{
    (void)in_sizes; (void)n_in; (void)out_size; (void)ws_size;

    const float* xin  = (const float*)d_in[0];
    const float* c1w = (const float*)d_in[1];  const float* c1b = (const float*)d_in[2];
    const float* b1g = (const float*)d_in[3];  const float* b1b = (const float*)d_in[4];
    const float* b1m = (const float*)d_in[5];  const float* b1v = (const float*)d_in[6];
    const float* c2w = (const float*)d_in[7];  const float* c2b = (const float*)d_in[8];
    const float* b2g = (const float*)d_in[9];  const float* b2b = (const float*)d_in[10];
    const float* b2m = (const float*)d_in[11]; const float* b2v = (const float*)d_in[12];
    const float* c3w = (const float*)d_in[13]; const float* c3b = (const float*)d_in[14];
    const float* b3g = (const float*)d_in[15]; const float* b3b = (const float*)d_in[16];
    const float* b3m = (const float*)d_in[17]; const float* b3v = (const float*)d_in[18];
    const float* ln_g = (const float*)d_in[19]; const float* ln_b = (const float*)d_in[20];
    const float* w_in = (const float*)d_in[21];
    const float* cw   = (const float*)d_in[22]; const float* cb   = (const float*)d_in[23];
    const float* w_x  = (const float*)d_in[24];
    const float* w_dt = (const float*)d_in[25]; const float* b_dt = (const float*)d_in[26];
    const float* a_log = (const float*)d_in[27]; const float* d_skip = (const float*)d_in[28];
    const float* w_out = (const float*)d_in[29];
    const float* fc_w = (const float*)d_in[30]; const float* fc_b = (const float*)d_in[31];

    float* ws = (float*)d_ws;
    float* bufA = ws;                       // 26,000,000 floats (conv outs / mamba scratch)
    float* bufB = ws + 26000000L;           //  6,500,000 floats (pool outs)
    float* bufC = ws + 32500000L;           //  7,000,000 floats (padded conv inputs)
    float* pooled = ws + 39500000L;         //      8,192 floats

    // mamba scratch carved inside bufA (stem use of bufA is finished by then)
    float* xt  = bufA;                      // tokens [B, L, ED]   1,605,632
    float* hln = bufA + 1605632L;           // [B*L, ED]           1,605,632
    float* xz  = bufA + 3211264L;           // [B*L, 1024]         6,422,528
    float* xc  = bufA + 9633792L;           // [B*L, 512]          3,211,264
    float* dbl = bufA + 12845056L;          // [B*L, 144]            903,168
    float* dtb = bufA + 13748224L;          // [B*L, 512]          3,211,264
    float* yb  = bufA + 16959488L;          // [B*L, 512]          3,211,264

    const int TOK = BATCH * SEQL;           // 6272

    // ---------------- conv stem (padded implicit-GEMM WMMA) ----------------
    // conv1: 3->64, 224->112 (s2 p1): K=27 (tail), N=64 (partial)
    {
        long pt = (long)BATCH * 3 * 226 * 226;
        pad_nchw_k<<<(pt + 255) / 256, 256, 0, stream>>>(xin, bufC, 3, 224, 224, 1);
        int M = BATCH * 112 * 112;
        conv3x3_bn_relu_wmma<true, true><<<dim3(M / 64, 1), 256, 0, stream>>>(
            bufC, c1w, bufA, 3, 226, 226, 64, 112, 112, 2, b1g, b1b, b1m, b1v, c1b);
        long tot = (long)BATCH * 64 * 56 * 56;
        maxpool_k<<<(tot + 255) / 256, 256, 0, stream>>>(bufA, bufB, 64, 112, 112, 56, 56, 3, 2, 1);
    }
    // conv2: 64->128, 56x56 (s1 p1): K=576, N=128 -> fully aligned
    {
        long pt = (long)BATCH * 64 * 58 * 58;
        pad_nchw_k<<<(pt + 255) / 256, 256, 0, stream>>>(bufB, bufC, 64, 56, 56, 1);
        int M = BATCH * 56 * 56;
        conv3x3_bn_relu_wmma<false, false><<<dim3(M / 64, 1), 256, 0, stream>>>(
            bufC, c2w, bufA, 64, 58, 58, 128, 56, 56, 1, b2g, b2b, b2m, b2v, c2b);
        long tot = (long)BATCH * 128 * 28 * 28;
        maxpool_k<<<(tot + 255) / 256, 256, 0, stream>>>(bufA, bufB, 128, 56, 56, 28, 28, 2, 2, 0);
    }
    // conv3: 128->256, 28x28 (s1 p1): K=1152, N=256 -> fully aligned
    {
        long pt = (long)BATCH * 128 * 30 * 30;
        pad_nchw_k<<<(pt + 255) / 256, 256, 0, stream>>>(bufB, bufC, 128, 28, 28, 1);
        int M = BATCH * 28 * 28;
        conv3x3_bn_relu_wmma<false, false><<<dim3(M / 64, 2), 256, 0, stream>>>(
            bufC, c3w, bufA, 128, 30, 30, 256, 28, 28, 1, b3g, b3b, b3m, b3v, c3b);
        long tot = (long)BATCH * 256 * 14 * 14;
        maxpool_k<<<(tot + 255) / 256, 256, 0, stream>>>(bufA, bufB, 256, 28, 28, 14, 14, 2, 2, 0);
    }
    // [B,256,196] -> tokens [B,196,256]
    {
        long tot = (long)BATCH * EDIM * SEQL;
        nchw_to_tokens_k<<<(tot + 255) / 256, 256, 0, stream>>>(bufB, xt);
    }

    // ---------------- mamba blocks ----------------
    for (int i = 0; i < NBLK; ++i) {
        layernorm_k<<<TOK, 256, 0, stream>>>(xt, ln_g + (long)i * EDIM, ln_b + (long)i * EDIM, hln);

        // xz = hln @ w_in^T        (M=6272, N=1024, K=256)  fully aligned
        gemm_bf16_wmma<false, false, false><<<dim3(TOK / 64, 8), 256, 0, stream>>>(
            hln, EDIM, w_in + (long)i * 2 * DIN * EDIM, EDIM,
            xz, 2 * DIN, nullptr, nullptr, TOK, 2 * DIN, EDIM, 0);

        // depthwise causal conv + silu -> xc
        {
            long tot = (long)TOK * DIN;
            dwconv_silu_k<<<(tot + 255) / 256, 256, 0, stream>>>(
                xz, cw + (long)i * DIN * DCONV, cb + (long)i * DIN, xc);
        }

        // dbl = xc @ w_x^T         (N=144 partial, K=512)
        gemm_bf16_wmma<false, true, false><<<dim3(TOK / 64, 2), 256, 0, stream>>>(
            xc, DIN, w_x + (long)i * (DRANK + 2 * DSTATE) * DIN, DIN,
            dbl, DRANK + 2 * DSTATE, nullptr, nullptr, TOK, DRANK + 2 * DSTATE, DIN, 0);

        // dt = softplus(dbl[:, :16] @ w_dt^T + b_dt)   (N=512, K=16 -> KG, uniform segs)
        gemm_bf16_wmma<false, false, true><<<dim3(TOK / 64, 4), 256, 0, stream>>>(
            dbl, DRANK + 2 * DSTATE, w_dt + (long)i * DIN * DRANK, DRANK,
            dtb, DIN, b_dt + (long)i * DIN, nullptr, TOK, DIN, DRANK, 1);

        // selective scan + gating -> yb
        ssm_scan_fused<<<dim3(DIN / 128, BATCH), 128, 0, stream>>>(
            dtb, dbl, xc, xz, a_log + (long)i * DIN * DSTATE, d_skip + (long)i * DIN, yb);

        // xt = yb @ w_out^T + xt   (N=256, K=512, residual)  fully aligned
        gemm_bf16_wmma<false, false, false><<<dim3(TOK / 64, 2), 256, 0, stream>>>(
            yb, DIN, w_out + (long)i * EDIM * DIN, DIN,
            xt, EDIM, nullptr, xt, TOK, EDIM, DIN, 0);
    }

    // ---------------- head ----------------
    meanpool_k<<<(BATCH * EDIM + 255) / 256, 256, 0, stream>>>(xt, pooled);
    gemm_bf16_wmma<true, true, false><<<dim3(1, 1), 256, 0, stream>>>(
        pooled, EDIM, fc_w, EDIM, (float*)d_out, NCLS, fc_b, nullptr, BATCH, NCLS, EDIM, 0);
}
